// COTREC_36077725286612
// MI455X (gfx1250) — compile-verified, matrix operand order checked by reading, not run on hardware
//
#include <hip/hip_runtime.h>
#include <hip/hip_bf16.h>

#define N_NODE  20000
#define EMB     100
#define BATCH   512
#define SEQ     50
#define LAYERS  2
#define KP      128          // K padded for bf16 WMMA (4 x 32)
#define TOPK    10
#define CLCHUNK 512          // columns staged in LDS per iteration
#define SIMLD   (CLCHUNK + 4) // padded LDS row stride: 516 mod 64 = 4 -> no bank conflicts
#define INV_TEMP 10.0f       // 1/0.1
#define W_K     10.0f
#define CL_W    100.0f

typedef __attribute__((ext_vector_type(16))) __bf16 v16bf;
typedef __attribute__((ext_vector_type(8)))  float  v8f;

// ---------------- WMMA helpers (CDNA5 wave32 layouts) ----------------

// A (16x32 bf16, MxK) fragment: lane -> m = lane%16, h = lane/16;
// halves 0..7 hold K = h*8 + i ; halves 8..15 hold K = 16 + h*8 + i.
// B (32x16, KxN) for C = X * Y^T with Y stored row-major [n][k] has the
// identical per-lane access pattern (n = lane%16), so one loader serves both.
__device__ inline v16bf load_frag_rm(const __bf16* M, int row0, int ld, int k0) {
    int lane = threadIdx.x & 31;
    int m = lane & 15;
    int h = lane >> 4;
    const __bf16* p = M + (size_t)(row0 + m) * ld + k0 + h * 8;
    v16bf a;
#pragma unroll
    for (int i = 0; i < 8; ++i) {
        a[i]     = p[i];
        a[8 + i] = p[16 + i];
    }
    return a;
}

__device__ inline v8f wmma_bf16(v16bf a, v16bf b, v8f c) {
    return __builtin_amdgcn_wmma_f32_16x16x32_bf16(
        /*neg_a=*/false, a, /*neg_b=*/false, b,
        /*c_mod=*/(short)0, c, /*reuse_a=*/false, /*reuse_b=*/false);
}

__device__ inline void topk_insert(float* topk, float v) {
    if (v > topk[TOPK - 1]) {
        int p = TOPK - 1;
        while (p > 0 && topk[p - 1] < v) { topk[p] = topk[p - 1]; --p; }
        topk[p] = v;
    }
}

// ---------------- generic small kernels ----------------

__global__ void k_fill(float* p, float v, int n) {
    int i = blockIdx.x * blockDim.x + threadIdx.x;
    if (i < n) p[i] = v;
}

__global__ void k_copy(const float* s, float* d, int n) {
    int i = blockIdx.x * blockDim.x + threadIdx.x;
    if (i < n) d[i] = s[i];
}

// ---------------- item_conv ----------------

__global__ void k_colsum(const float* vals, const int* col, float* colsum, int E) {
    int e = blockIdx.x * blockDim.x + threadIdx.x;
    if (e < E) atomicAdd(&colsum[col[e]], vals[e]);
}

__global__ void k_normvals(const float* vals, const int* col, const float* colsum,
                           float* out, int E) {
    int e = blockIdx.x * blockDim.x + threadIdx.x;
    if (e < E) out[e] = vals[e] / colsum[col[e]];
}

__global__ void k_att_logits(const float* x, const float* aw, const float* ab, float* lg) {
    int n = blockIdx.x * blockDim.x + threadIdx.x;
    if (n >= N_NODE) return;
    const float* r = x + (size_t)n * EMB;
    float s = ab[0];
    for (int k = 0; k < EMB; ++k) s += r[k] * aw[k];
    lg[n] = s;
}

// single-block softmax stats over N_NODE values: stats[0]=max, stats[1]=sum exp
__global__ void k_softmax_stats(const float* lg, float* stats) {
    __shared__ float red[1024];
    int t = threadIdx.x;
    float mx = -3.0e38f;
    for (int i = t; i < N_NODE; i += 1024) mx = fmaxf(mx, lg[i]);
    red[t] = mx; __syncthreads();
    for (int s = 512; s > 0; s >>= 1) {
        if (t < s) red[t] = fmaxf(red[t], red[t + s]);
        __syncthreads();
    }
    mx = red[0]; __syncthreads();
    float se = 0.f;
    for (int i = t; i < N_NODE; i += 1024) se += __expf(lg[i] - mx);
    red[t] = se; __syncthreads();
    for (int s = 512; s > 0; s >>= 1) {
        if (t < s) red[t] += red[t + s];
        __syncthreads();
    }
    if (t == 0) { stats[0] = mx; stats[1] = red[0]; }
}

// xW[n][o] = sum_k x[n][k] * w[o][k]   (x @ W^T)
__global__ void k_xw(const float* x, const float* w, float* xw) {
    int i = blockIdx.x * blockDim.x + threadIdx.x;
    if (i >= N_NODE * EMB) return;
    int n = i / EMB, o = i % EMB;
    const float* r  = x + (size_t)n * EMB;
    const float* wr = w + (size_t)o * EMB;
    float s = 0.f;
    for (int k = 0; k < EMB; ++k) s += r[k] * wr[k];
    xw[i] = s;
}

__global__ void k_edge_scatter(const int* rowv, const int* colv, const float* vals_n,
                               const float* xw, const float* lg, const float* stats,
                               float* xnew, int E) {
    int e = blockIdx.x * blockDim.x + threadIdx.x;
    if (e >= E) return;
    int c = colv[e], r = rowv[e];
    float att = __expf(lg[c] - stats[0]) / stats[1];
    float s = vals_n[e] * att;
    const float* src = xw + (size_t)c * EMB;
    float* dst = xnew + (size_t)r * EMB;
#pragma unroll 4
    for (int d = 0; d < EMB; ++d) atomicAdd(dst + d, s * src[d]);
}

__global__ void k_l2_update(const float* xnew, float* x, float* fin) {
    int n = blockIdx.x * blockDim.x + threadIdx.x;
    if (n >= N_NODE) return;
    const float* r = xnew + (size_t)n * EMB;
    float s = 0.f;
    for (int k = 0; k < EMB; ++k) s += r[k] * r[k];
    float inv = 1.f / fmaxf(sqrtf(s), 1e-12f);
    float* xo = x + (size_t)n * EMB;
    float* fo = fin + (size_t)n * EMB;
    for (int k = 0; k < EMB; ++k) {
        float v = r[k] * inv;
        xo[k] = v;
        fo[k] += v;
    }
}

// final /= 3 (-> item_emb, in place), then l2norm -> bf16 padded [N][KP]
__global__ void k_item_final(float* fin, __bf16* itemn) {
    int n = blockIdx.x * blockDim.x + threadIdx.x;
    if (n >= N_NODE) return;
    float* r = fin + (size_t)n * EMB;
    float s = 0.f;
    for (int k = 0; k < EMB; ++k) {
        float v = r[k] * (1.0f / 3.0f);
        r[k] = v;
        s += v * v;
    }
    float inv = 1.f / fmaxf(sqrtf(s), 1e-12f);
    __bf16* o = itemn + (size_t)n * KP;
    for (int k = 0; k < EMB; ++k) o[k] = (__bf16)(r[k] * inv);
    for (int k = EMB; k < KP; ++k)  o[k] = (__bf16)0.0f;
}

// ---------------- session embedding ----------------

__global__ void k_gather(const int* rsi, const float* item_emb, float* seq_h) {
    int i = blockIdx.x * blockDim.x + threadIdx.x;
    if (i >= BATCH * SEQ * EMB) return;
    int d = i % EMB;
    int l = (i / EMB) % SEQ;
    int b = i / (EMB * SEQ);
    int id = rsi[b * SEQ + l];
    seq_h[i] = (id == 0) ? 0.f : item_emb[(size_t)(id - 1) * EMB + d];
}

__global__ void k_hs(const float* seq_h, const float* slen, float* hs) {
    int i = blockIdx.x * blockDim.x + threadIdx.x;
    if (i >= BATCH * EMB) return;
    int d = i % EMB, b = i / EMB;
    float s = 0.f;
    for (int l = 0; l < SEQ; ++l) s += seq_h[((size_t)b * SEQ + l) * EMB + d];
    hs[i] = s / slen[b];
}

// hsg = hs @ glu2_w^T
__global__ void k_hsg(const float* hs, const float* glu2_w, float* hsg) {
    int i = blockIdx.x * blockDim.x + threadIdx.x;
    if (i >= BATCH * EMB) return;
    int o = i % EMB, b = i / EMB;
    const float* r = hs + (size_t)b * EMB;
    const float* w = glu2_w + (size_t)o * EMB;
    float s = 0.f;
    for (int k = 0; k < EMB; ++k) s += r[k] * w[k];
    hsg[i] = s;
}

// nh1 = tanh([pos || seq_h] @ w_1), w_1: [2E][E]
__global__ void k_nh1(const float* seq_h, const float* pos, const float* w1, float* nh1) {
    int i = blockIdx.x * blockDim.x + threadIdx.x;
    if (i >= BATCH * SEQ * EMB) return;
    int o = i % EMB;
    int l = (i / EMB) % SEQ;
    int bl = i / EMB;
    const float* pr = pos + (size_t)l * EMB;
    const float* sr = seq_h + (size_t)bl * EMB;
    float s = 0.f;
    for (int k = 0; k < EMB; ++k) s += pr[k] * w1[(size_t)k * EMB + o];
    for (int k = 0; k < EMB; ++k) s += sr[k] * w1[(size_t)(EMB + k) * EMB + o];
    nh1[i] = tanhf(s);
}

// nh2 = sigmoid(nh1 @ glu1_w^T + glu1_b + hsg[b])
__global__ void k_nh2(const float* nh1, const float* glu1_w, const float* glu1_b,
                      const float* hsg, float* nh2) {
    int i = blockIdx.x * blockDim.x + threadIdx.x;
    if (i >= BATCH * SEQ * EMB) return;
    int o = i % EMB;
    int bl = i / EMB;
    int b = bl / SEQ;
    const float* r = nh1 + (size_t)bl * EMB;
    const float* w = glu1_w + (size_t)o * EMB;
    float s = glu1_b[o] + hsg[(size_t)b * EMB + o];
    for (int k = 0; k < EMB; ++k) s += r[k] * w[k];
    nh2[i] = 1.f / (1.f + __expf(-s));
}

__global__ void k_beta(const float* nh2, const float* w2, const int* mask, float* beta) {
    int i = blockIdx.x * blockDim.x + threadIdx.x;
    if (i >= BATCH * SEQ) return;
    const float* r = nh2 + (size_t)i * EMB;
    float s = 0.f;
    for (int k = 0; k < EMB; ++k) s += r[k] * w2[k];
    beta[i] = s * (float)mask[i];
}

__global__ void k_sess(const float* beta, const float* seq_h, float* sess) {
    int i = blockIdx.x * blockDim.x + threadIdx.x;
    if (i >= BATCH * EMB) return;
    int d = i % EMB, b = i / EMB;
    float s = 0.f;
    for (int l = 0; l < SEQ; ++l)
        s += beta[b * SEQ + l] * seq_h[((size_t)b * SEQ + l) * EMB + d];
    sess[i] = s;
}

// sess -> W_K * l2norm(sess), store as bf16 padded [BATCH][KP]
__global__ void k_sess_norm(const float* sess, __bf16* sess_bf) {
    int b = blockIdx.x * blockDim.x + threadIdx.x;
    if (b >= BATCH) return;
    const float* r = sess + (size_t)b * EMB;
    float s = 0.f;
    for (int k = 0; k < EMB; ++k) s += r[k] * r[k];
    float inv = W_K / fmaxf(sqrtf(s), 1e-12f);
    __bf16* o = sess_bf + (size_t)b * KP;
    for (int k = 0; k < EMB; ++k) o[k] = (__bf16)(r[k] * inv);
    for (int k = EMB; k < KP; ++k)  o[k] = (__bf16)0.0f;
}

// ---------------- WMMA matmuls ----------------

// scores[m][n] = sess_bf[m][:] . itemn_bf[n][:]   (one wave per 16x16 tile)
__global__ void k_scores_wmma(const __bf16* sess_bf, const __bf16* itemn_bf, float* out) {
    int ntile = blockIdx.x;   // 0..1249
    int mtile = blockIdx.y;   // 0..31
    v8f c = {};
#pragma unroll
    for (int kc = 0; kc < 4; ++kc) {
        v16bf a = load_frag_rm(sess_bf, mtile * 16, KP, kc * 32);
        v16bf b = load_frag_rm(itemn_bf, ntile * 16, KP, kc * 32);
        c = wmma_bf16(a, b, c);
    }
    int lane = threadIdx.x & 31;
    int nn = lane & 15;
    int h = lane >> 4;
#pragma unroll
    for (int r = 0; r < 8; ++r) {
        int m = r + h * 8;
        out[(size_t)(mtile * 16 + m) * N_NODE + ntile * 16 + nn] = c[r];
    }
}

// fused N x N similarity (WMMA) + per-row top-10 + CE(label=0) accumulation.
// 16 rows per workgroup; 4 waves do WMMA tiles, all 128 threads scan
// (8 scanner threads per row, private top-10 each, merged at the end).
__global__ void __launch_bounds__(128)
k_cl_topk(const __bf16* itemn_bf, float* cl_acc) {
    __shared__ float sim[16 * SIMLD];     // ~33 KB, padded rows (bank-conflict free)
    int rowtile = blockIdx.x;             // 0..1249
    int wave = threadIdx.x >> 5;
    int lane = threadIdx.x & 31;
    // A fragments for these 16 rows, all K chunks (reused every column tile)
    v16bf afr[4];
#pragma unroll
    for (int kc = 0; kc < 4; ++kc)
        afr[kc] = load_frag_rm(itemn_bf, rowtile * 16, KP, kc * 32);

    float topk[TOPK];
#pragma unroll
    for (int i = 0; i < TOPK; ++i) topk[i] = -3.0e38f;

    int nn = lane & 15;
    int h = lane >> 4;
    int scan_r = threadIdx.x >> 3;        // row owned for scanning (0..15)
    int scan_s = threadIdx.x & 7;         // scanner id within row (0..7)
    const int nchunks = (N_NODE + CLCHUNK - 1) / CLCHUNK;   // 40
    for (int cch = 0; cch < nchunks; ++cch) {
        int colbase = cch * CLCHUNK;
        int ncols = min(CLCHUNK, N_NODE - colbase);         // multiple of 16
        int ntiles = ncols / 16;
        for (int t = wave; t < ntiles; t += 4) {
            // prefetch this wave's next B tile (stays hot in L0/L2 behind WMMA)
            int tn = t + 4;
            if (tn < ntiles) {
                const __bf16* pf =
                    itemn_bf + (size_t)(colbase + tn * 16 + nn) * KP + h * 8;
                __builtin_prefetch(pf, 0, 3);
            }
            v8f c = {};
#pragma unroll
            for (int kc = 0; kc < 4; ++kc) {
                v16bf b = load_frag_rm(itemn_bf, colbase + t * 16, KP, kc * 32);
                c = wmma_bf16(afr[kc], b, c);
            }
#pragma unroll
            for (int r = 0; r < 8; ++r) {
                int m = r + h * 8;
                sim[m * SIMLD + t * 16 + nn] = c[r] * INV_TEMP;
            }
        }
        __syncthreads();
        {
            const float* sr = sim + scan_r * SIMLD;
            for (int j = scan_s; j < ncols; j += 8)
                topk_insert(topk, sr[j]);
        }
        __syncthreads();
    }
    // merge 8 candidate lists per row (reuse sim LDS: 16*8*TOPK = 1280 floats)
    {
        float* cand = sim;
#pragma unroll
        for (int i = 0; i < TOPK; ++i)
            cand[(scan_r * 8 + scan_s) * TOPK + i] = topk[i];
    }
    __syncthreads();
    if (threadIdx.x < 16) {
        int r = threadIdx.x;
        float best[TOPK];
#pragma unroll
        for (int i = 0; i < TOPK; ++i) best[i] = -3.0e38f;
        const float* c = sim + r * 8 * TOPK;
        for (int j = 0; j < 8 * TOPK; ++j)
            topk_insert(best, c[j]);
        float m0 = best[0];
        float s = 0.f;
#pragma unroll
        for (int i = 0; i < TOPK; ++i) s += __expf(best[i] - m0);
        // loss_row = logsumexp(top10) - top10[0] = log(s)
        atomicAdd(cl_acc, logf(s));
    }
}

// ---------------- losses ----------------

__global__ void k_loss_item(const float* scores, const int* tar, float* acc) {
    __shared__ float red[256];
    int b = blockIdx.x;
    const float* row = scores + (size_t)b * N_NODE;
    int t = threadIdx.x;
    float mx = -3.0e38f;
    for (int j = t; j < N_NODE; j += 256) mx = fmaxf(mx, row[j]);
    red[t] = mx; __syncthreads();
    for (int s = 128; s > 0; s >>= 1) {
        if (t < s) red[t] = fmaxf(red[t], red[t + s]);
        __syncthreads();
    }
    mx = red[0]; __syncthreads();
    float se = 0.f;
    for (int j = t; j < N_NODE; j += 256) se += __expf(row[j] - mx);
    red[t] = se; __syncthreads();
    for (int s = 128; s > 0; s >>= 1) {
        if (t < s) red[t] += red[t + s];
        __syncthreads();
    }
    if (t == 0) {
        float lse = mx + logf(red[0]);
        atomicAdd(acc, lse - row[tar[b]]);
    }
}

__global__ void k_finalize(const float* acc, float* out) {
    out[0] = acc[0] / (float)BATCH;
    out[1 + (size_t)BATCH * N_NODE] = CL_W * acc[1] / (float)N_NODE;
}

// ---------------- host driver ----------------

extern "C" void kernel_launch(void* const* d_in, const int* in_sizes, int n_in,
                              void* d_out, int out_size, void* d_ws, size_t ws_size,
                              hipStream_t stream) {
    (void)n_in; (void)out_size; (void)ws_size;
    const float* session_len = (const float*)d_in[1];
    const int*   rsi         = (const int*)d_in[2];
    const int*   maskp       = (const int*)d_in[3];
    const int*   tar         = (const int*)d_in[5];
    const int*   adj_row     = (const int*)d_in[8];
    const int*   adj_col     = (const int*)d_in[9];
    const float* adj_vals    = (const float*)d_in[10];
    const float* embedding   = (const float*)d_in[11];
    const float* pos_emb     = (const float*)d_in[12];
    const float* w_item      = (const float*)d_in[13];
    const float* att_w       = (const float*)d_in[14];
    const float* att_b       = (const float*)d_in[15];
    const float* w_1         = (const float*)d_in[16];
    const float* w_2         = (const float*)d_in[17];
    const float* glu1_w      = (const float*)d_in[18];
    const float* glu1_b      = (const float*)d_in[19];
    const float* glu2_w      = (const float*)d_in[20];
    const int E = in_sizes[8];

    // workspace carve-up
    char* wp = (char*)d_ws;
    auto carve = [&](size_t bytes) {
        char* p = wp;
        wp += (bytes + 255) & ~(size_t)255;
        return p;
    };
    float* colsum  = (float*)carve(sizeof(float) * N_NODE);
    float* vals_n  = (float*)carve(sizeof(float) * E);
    float* x       = (float*)carve(sizeof(float) * N_NODE * EMB);
    float* xw      = (float*)carve(sizeof(float) * N_NODE * EMB);
    float* xnew    = (float*)carve(sizeof(float) * N_NODE * EMB);
    float* fin     = (float*)carve(sizeof(float) * N_NODE * EMB);  // -> item_emb
    float* lg      = (float*)carve(sizeof(float) * N_NODE);
    float* stats   = (float*)carve(sizeof(float) * 4);
    float* acc     = (float*)carve(sizeof(float) * 4);
    __bf16* itemn_bf = (__bf16*)carve(sizeof(__bf16) * (size_t)N_NODE * KP);
    __bf16* sess_bf  = (__bf16*)carve(sizeof(__bf16) * (size_t)BATCH * KP);
    float* seq_h   = (float*)carve(sizeof(float) * (size_t)BATCH * SEQ * EMB);
    float* hs      = (float*)carve(sizeof(float) * BATCH * EMB);
    float* hsg     = (float*)carve(sizeof(float) * BATCH * EMB);
    float* nh1     = (float*)carve(sizeof(float) * (size_t)BATCH * SEQ * EMB);
    float* nh2     = (float*)carve(sizeof(float) * (size_t)BATCH * SEQ * EMB);
    float* betab   = (float*)carve(sizeof(float) * BATCH * SEQ);
    float* sessv   = (float*)carve(sizeof(float) * BATCH * EMB);

    auto nb = [](long n, int b) { return (int)((n + b - 1) / b); };
    const int NE = N_NODE * EMB;
    const int BLE = BATCH * SEQ * EMB;

    // ---- graph conv ----
    k_fill<<<nb(N_NODE, 256), 256, 0, stream>>>(colsum, 0.f, N_NODE);
    k_colsum<<<nb(E, 256), 256, 0, stream>>>(adj_vals, adj_col, colsum, E);
    k_normvals<<<nb(E, 256), 256, 0, stream>>>(adj_vals, adj_col, colsum, vals_n, E);
    k_copy<<<nb(NE, 256), 256, 0, stream>>>(embedding, x, NE);
    k_copy<<<nb(NE, 256), 256, 0, stream>>>(embedding, fin, NE);
    for (int layer = 0; layer < LAYERS; ++layer) {
        k_att_logits<<<nb(N_NODE, 256), 256, 0, stream>>>(x, att_w, att_b, lg);
        k_softmax_stats<<<1, 1024, 0, stream>>>(lg, stats);
        k_xw<<<nb(NE, 256), 256, 0, stream>>>(x, w_item + (size_t)layer * EMB * EMB, xw);
        k_fill<<<nb(NE, 256), 256, 0, stream>>>(xnew, 0.f, NE);
        k_edge_scatter<<<nb(E, 256), 256, 0, stream>>>(adj_row, adj_col, vals_n, xw, lg,
                                                       stats, xnew, E);
        k_l2_update<<<nb(N_NODE, 256), 256, 0, stream>>>(xnew, x, fin);
    }
    k_item_final<<<nb(N_NODE, 256), 256, 0, stream>>>(fin, itemn_bf);

    // ---- session embedding ----
    k_gather<<<nb(BLE, 256), 256, 0, stream>>>(rsi, fin, seq_h);
    k_hs<<<nb(BATCH * EMB, 256), 256, 0, stream>>>(seq_h, session_len, hs);
    k_hsg<<<nb(BATCH * EMB, 256), 256, 0, stream>>>(hs, glu2_w, hsg);
    k_nh1<<<nb(BLE, 256), 256, 0, stream>>>(seq_h, pos_emb, w_1, nh1);
    k_nh2<<<nb(BLE, 256), 256, 0, stream>>>(nh1, glu1_w, glu1_b, hsg, nh2);
    k_beta<<<nb(BATCH * SEQ, 256), 256, 0, stream>>>(nh2, w_2, maskp, betab);
    k_sess<<<nb(BATCH * EMB, 256), 256, 0, stream>>>(betab, seq_h, sessv);
    k_sess_norm<<<nb(BATCH, 256), 256, 0, stream>>>(sessv, sess_bf);

    // ---- scores (WMMA) + losses ----
    float* out = (float*)d_out;
    float* scores = out + 1;
    k_fill<<<1, 32, 0, stream>>>(acc, 0.f, 4);
    {
        dim3 grid(N_NODE / 16, BATCH / 16);   // 1250 x 32 waves
        k_scores_wmma<<<grid, 32, 0, stream>>>(sess_bf, itemn_bf, scores);
    }
    k_loss_item<<<BATCH, 256, 0, stream>>>(scores, tar, &acc[0]);
    k_cl_topk<<<N_NODE / 16, 128, 0, stream>>>(itemn_bf, &acc[1]);
    k_finalize<<<1, 1, 0, stream>>>(acc, out);
}
